// RegionProposer_19069654794836
// MI455X (gfx1250) — matrix operand order repaired and checked by reading, not compiled, not run on hardware
//
#include <hip/hip_runtime.h>
#include <hip/hip_bf16.h>

typedef __attribute__((ext_vector_type(16))) _Float16 v16h;
typedef __attribute__((ext_vector_type(8)))  _Float16 v8h;
typedef __attribute__((ext_vector_type(8)))  float    v8f;

#define N_NODES 50000
#define N_EDGES 800000
#define E_TOT   (N_EDGES + N_NODES)   // self-loops appended
#define HFDIM   256                   // HEADS*HIDDEN
#define HIDDEN  64
#define HEADS   4
#define CTXDIM  32
#define NCAND   8192
#define NBUCK   7

// ---------- helpers ----------
static __device__ __forceinline__ unsigned f2ord(float f) {
  unsigned u = __float_as_uint(f);
  return (u & 0x80000000u) ? ~u : (u | 0x80000000u);   // order-preserving map
}
static __device__ __forceinline__ float ord2f(unsigned u) {
  return (u & 0x80000000u) ? __uint_as_float(u & 0x7FFFFFFFu) : __uint_as_float(~u);
}
static __device__ __forceinline__ void edge_sd(const int* __restrict__ ei, int eid,
                                               int& s, int& d) {
  if (eid < N_EDGES) { s = ei[eid]; d = ei[N_EDGES + eid]; }
  else               { s = eid - N_EDGES; d = s; }       // self-loop
}

// ---------- fp32 -> f16 conversion kernels ----------
// pad columns: x [rows,cin] -> [rows,cout] f16, zero-pad tail columns
__global__ void pad_cols_f16(const float* __restrict__ in, _Float16* __restrict__ out,
                             int rows, int cin, int cout) {
  int i = blockIdx.x * blockDim.x + threadIdx.x;
  if (i >= rows * cout) return;
  int r = i / cout, c = i - r * cout;
  out[i] = (c < cin) ? (_Float16)in[(size_t)r * cin + c] : (_Float16)0.0f;
}
// transpose+pad: W [rin,256] row-major -> Wt [256,Krnd] (column-major of padded W)
__global__ void transpose_pad_f16(const float* __restrict__ in, _Float16* __restrict__ out,
                                  int rin, int Krnd, int cols) {
  int i = blockIdx.x * blockDim.x + threadIdx.x;
  if (i >= cols * Krnd) return;
  int c = i / Krnd, k = i - c * Krnd;
  out[i] = (k < rin) ? (_Float16)in[(size_t)k * cols + c] : (_Float16)0.0f;
}

// ---------- WMMA GEMM: C[M,256] = A[M,K] * B[K,256] ----------
// A row-major [M,K] f16; Bt = B^T col-major [256,K] f16.
// One wave computes a 16x64 output strip (4 WMMA tiles sharing the A fragment).
template <int K>
__global__ __launch_bounds__(128) void wmma_gemm(
    const _Float16* __restrict__ A, const _Float16* __restrict__ Bt,
    float* __restrict__ C, int M) {
  const int lane  = threadIdx.x & 31;
  const int wtile = blockIdx.x * 4 + (threadIdx.x >> 5);
  const int total = (M / 16) * (HFDIM / 64);
  if (wtile >= total) return;                 // uniform per wave (EXEC stays all-ones)
  const int tm = wtile >> 2;                  // 16-row tile
  const int tg = wtile & 3;                   // 64-col group

  const int row  = tm * 16 + (lane & 15);
  const int coln = lane & 15;
  const int koA  = (lane < 16) ? 0 : 16;      // A: lanes 0-15 K0..15,      lanes 16-31 K16..31
  const int koB  = (lane < 16) ? 0 : 16;      // B: lanes 0-15 K0..15,      lanes 16-31 K16..31

  v8f c0 = {}, c1 = {}, c2 = {}, c3 = {};
#pragma unroll
  for (int kb = 0; kb < K; kb += 32) {
    // A fragment: two contiguous 16B vector loads, packed to v16h
    const v8h* Ap = (const v8h*)(A + (size_t)row * K + kb);
    v8h a0 = Ap[koA >> 3];
    v8h a1 = Ap[1 + (koA >> 3)];
    v16h a = __builtin_shufflevector(a0, a1, 0, 1, 2, 3, 4, 5, 6, 7,
                                     8, 9, 10, 11, 12, 13, 14, 15);
    // B fragments: 16 contiguous halves per lane from transposed weights
#pragma unroll
    for (int j = 0; j < 4; ++j) {
      const int col = tg * 64 + j * 16 + coln;
      const v8h* Bp = (const v8h*)(Bt + (size_t)col * K + kb + koB);
      v8h b0 = Bp[0];
      v8h b1 = Bp[1];
      v16h b = __builtin_shufflevector(b0, b1, 0, 1, 2, 3, 4, 5, 6, 7,
                                       8, 9, 10, 11, 12, 13, 14, 15);
      v8f& cj = (j == 0) ? c0 : (j == 1) ? c1 : (j == 2) ? c2 : c3;
      cj = __builtin_amdgcn_wmma_f32_16x16x32_f16(false, a, false, b, (short)0, cj,
                                                  false, false);
    }
  }
  const int rowc = tm * 16 + ((lane < 16) ? 0 : 8);
  float* Cp = C + (size_t)rowc * HFDIM + tg * 64 + coln;
#pragma unroll
  for (int r = 0; r < 8; ++r) {
    Cp[(size_t)r * HFDIM + 0]  = c0[r];
    Cp[(size_t)r * HFDIM + 16] = c1[r];
    Cp[(size_t)r * HFDIM + 32] = c2[r];
    Cp[(size_t)r * HFDIM + 48] = c3[r];
  }
}

// ---------- attention coefficients: a_src/a_dst [N,4] ----------
__global__ void att_coef(const float* __restrict__ hfeat,
                         const float* __restrict__ as, const float* __restrict__ ad,
                         float* __restrict__ a_src, float* __restrict__ a_dst) {
  int i = blockIdx.x * blockDim.x + threadIdx.x;
  if (i >= N_NODES * HEADS) return;
  int n = i >> 2, h = i & 3;
  const float* hp  = hfeat + (size_t)n * HFDIM + h * HIDDEN;
  const float* asp = as + h * HIDDEN;
  const float* adp = ad + h * HIDDEN;
  float s = 0.f, d = 0.f;
#pragma unroll 8
  for (int f = 0; f < HIDDEN; ++f) { float v = hp[f]; s += v * asp[f]; d += v * adp[f]; }
  a_src[i] = s; a_dst[i] = d;
}

// ---------- segment-softmax pass 1: per-dst max ----------
__global__ void edge_max(const int* __restrict__ ei,
                         const float* __restrict__ a_src, const float* __restrict__ a_dst,
                         unsigned* __restrict__ nmax) {
  int i = blockIdx.x * blockDim.x + threadIdx.x;
  if (i >= E_TOT * HEADS) return;
  int eid = i >> 2, h = i & 3, s, d;
  edge_sd(ei, eid, s, d);
  float e = a_src[s * 4 + h] + a_dst[d * 4 + h];
  e = (e >= 0.f) ? e : 0.2f * e;
  atomicMax(&nmax[d * 4 + h], f2ord(e));
}

// ---------- pass 2: per-dst sum of exp; cache exp per (edge,head) ----------
__global__ void edge_sum(const int* __restrict__ ei,
                         const float* __restrict__ a_src, const float* __restrict__ a_dst,
                         const unsigned* __restrict__ nmax, float* __restrict__ nsum,
                         float* __restrict__ exbuf) {
  int i = blockIdx.x * blockDim.x + threadIdx.x;
  if (i >= E_TOT * HEADS) return;
  int eid = i >> 2, h = i & 3, s, d;
  edge_sd(ei, eid, s, d);
  float e = a_src[s * 4 + h] + a_dst[d * 4 + h];
  e = (e >= 0.f) ? e : 0.2f * e;
  float m = ord2f(nmax[d * 4 + h]);
  float ex = __expf(e - m);
  exbuf[i] = ex;
  atomicAdd(&nsum[d * 4 + h], ex);
}

// ---------- pass 3: weighted message scatter (one block per edge) ----------
__global__ __launch_bounds__(256) void edge_msg(const int* __restrict__ ei,
    const float* __restrict__ nsum, const float* __restrict__ exbuf,
    const float* __restrict__ hfeat, float* __restrict__ acc) {
  int eid = blockIdx.x;
  int t = threadIdx.x;          // t = h*64 + f
  int h = t >> 6;
  int s, d;
  edge_sd(ei, eid, s, d);
  float alpha = exbuf[eid * 4 + h] / nsum[d * 4 + h];
  float v = hfeat[(size_t)s * HFDIM + t] * alpha;
  atomicAdd(&acc[(size_t)d * HFDIM + t], v);
}

// ---------- mean over heads + bias + ELU; also emit f16 for next GEMM ----------
__global__ void finalize_layer(const float* __restrict__ acc, const float* __restrict__ bias,
                               float* __restrict__ out32, _Float16* __restrict__ out16) {
  int i = blockIdx.x * blockDim.x + threadIdx.x;
  if (i >= N_NODES * HIDDEN) return;
  int n = i >> 6, f = i & 63;
  const float* ap = acc + (size_t)n * HFDIM + f;
  float v = 0.25f * (ap[0] + ap[64] + ap[128] + ap[192]) + bias[f];
  v = (v > 0.f) ? v : (__expf(v) - 1.0f);   // ELU
  out32[i] = v;
  out16[i] = (_Float16)v;
}

// ---------- start-logit MLP: one block (64 threads) per candidate ----------
__global__ __launch_bounds__(64) void start_mlp(const float* __restrict__ h2,
    const float* __restrict__ donor, const int* __restrict__ cand,
    const float* __restrict__ sW1, const float* __restrict__ sb1,
    const float* __restrict__ sW2, const float* __restrict__ sb2,
    float* __restrict__ out) {
  int i = blockIdx.x, j = threadIdx.x;
  int node = cand[i];
  const float* hp = h2 + (size_t)node * HIDDEN;
  float a = sb1[j];
#pragma unroll 4
  for (int k = 0; k < HIDDEN; ++k) a += hp[k] * sW1[k * HIDDEN + j];
#pragma unroll 4
  for (int k = 0; k < CTXDIM; ++k) a += donor[k] * sW1[(HIDDEN + k) * HIDDEN + j];
  a = fmaxf(a, 0.f) * sW2[j];
  __shared__ float red[64];
  red[j] = a; __syncthreads();
  for (int s = 32; s > 0; s >>= 1) { if (j < s) red[j] += red[j + s]; __syncthreads(); }
  if (j == 0) out[i] = red[0] + sb2[0];
}

// ---------- pooled ctx + length-logit MLP (single block) ----------
__global__ __launch_bounds__(128) void length_mlp(const float* __restrict__ h2,
    const float* __restrict__ donor, const int* __restrict__ cand,
    const float* __restrict__ lW1, const float* __restrict__ lb1,
    const float* __restrict__ lW2, const float* __restrict__ lb2,
    float* __restrict__ out) {
  __shared__ float pooled[HIDDEN + CTXDIM];
  __shared__ float hid[HIDDEN];
  int t = threadIdx.x;
  if (t < HIDDEN + CTXDIM) {
    float s;
    if (t < HIDDEN) {
      s = 0.f;
      for (int i = 0; i < NCAND; ++i) s += h2[(size_t)cand[i] * HIDDEN + t];
      s *= (1.0f / NCAND);
    } else {
      s = donor[t - HIDDEN];            // mean of a broadcast row is the row
    }
    pooled[t] = s;
  }
  __syncthreads();
  if (t < HIDDEN) {
    float a = lb1[t];
    for (int k = 0; k < HIDDEN + CTXDIM; ++k) a += pooled[k] * lW1[k * HIDDEN + t];
    hid[t] = fmaxf(a, 0.f);
  }
  __syncthreads();
  if (t < NBUCK) {
    float a = lb2[t];
    for (int j = 0; j < HIDDEN; ++j) a += hid[j] * lW2[j * NBUCK + t];
    out[NCAND + t] = a;
  }
}

// =================== host side ===================
extern "C" void kernel_launch(void* const* d_in, const int* in_sizes, int n_in,
                              void* d_out, int out_size, void* d_ws, size_t ws_size,
                              hipStream_t stream) {
  const float* x     = (const float*)d_in[0];
  const int*   ei    = (const int*)d_in[1];
  const float* donor = (const float*)d_in[2];
  const int*   cand  = (const int*)d_in[3];
  const float* W1  = (const float*)d_in[4];
  const float* as1 = (const float*)d_in[5];
  const float* ad1 = (const float*)d_in[6];
  const float* b1  = (const float*)d_in[7];
  const float* W2  = (const float*)d_in[8];
  const float* as2 = (const float*)d_in[9];
  const float* ad2 = (const float*)d_in[10];
  const float* b2  = (const float*)d_in[11];
  const float* sW1 = (const float*)d_in[12];
  const float* sb1 = (const float*)d_in[13];
  const float* sW2 = (const float*)d_in[14];
  const float* sb2 = (const float*)d_in[15];
  const float* lW1 = (const float*)d_in[16];
  const float* lb1 = (const float*)d_in[17];
  const float* lW2 = (const float*)d_in[18];
  const float* lb2 = (const float*)d_in[19];
  float* out = (float*)d_out;

  // workspace carve-out (~139 MB)
  float*    h_feat = (float*)d_ws;                       // [N,256]
  float*    acc    = h_feat + (size_t)N_NODES * HFDIM;   // [N,256]
  float*    h_out  = acc    + (size_t)N_NODES * HFDIM;   // [N,64]
  float*    a_src  = h_out  + (size_t)N_NODES * HIDDEN;  // [N,4]
  float*    a_dst  = a_src  + (size_t)N_NODES * HEADS;
  unsigned* nmax   = (unsigned*)(a_dst + (size_t)N_NODES * HEADS);
  float*    nsum   = (float*)(nmax + (size_t)N_NODES * HEADS);
  float*    exbuf  = nsum + (size_t)N_NODES * HEADS;     // [E_TOT,4]
  _Float16* ah     = (_Float16*)(exbuf + (size_t)E_TOT * HEADS); // [N,64] f16 (also [N,32])
  _Float16* wt     = ah + (size_t)N_NODES * HIDDEN;      // [256,64] f16 transposed weights

  const int TPB = 256;
  const int gemm_blocks = (N_NODES / 16) * (HFDIM / 64) / 4;      // 4 waves/block, 16x64/wave
  const int att_blocks  = (N_NODES * HEADS + TPB - 1) / TPB;
  const int eh_blocks   = (E_TOT * HEADS + TPB - 1) / TPB;
  const int fin_blocks  = (N_NODES * HIDDEN + TPB - 1) / TPB;

  // ---------- layer 1 ----------
  pad_cols_f16<<<(N_NODES * 32 + TPB - 1) / TPB, TPB, 0, stream>>>(x, ah, N_NODES, 27, 32);
  transpose_pad_f16<<<(HFDIM * 32 + TPB - 1) / TPB, TPB, 0, stream>>>(W1, wt, 27, 32, HFDIM);
  wmma_gemm<32><<<gemm_blocks, 128, 0, stream>>>(ah, wt, h_feat, N_NODES);
  att_coef<<<att_blocks, TPB, 0, stream>>>(h_feat, as1, ad1, a_src, a_dst);
  hipMemsetAsync(nmax, 0, (size_t)N_NODES * HEADS * sizeof(unsigned), stream);
  hipMemsetAsync(nsum, 0, (size_t)N_NODES * HEADS * sizeof(float), stream);
  hipMemsetAsync(acc,  0, (size_t)N_NODES * HFDIM * sizeof(float), stream);
  edge_max<<<eh_blocks, TPB, 0, stream>>>(ei, a_src, a_dst, nmax);
  edge_sum<<<eh_blocks, TPB, 0, stream>>>(ei, a_src, a_dst, nmax, nsum, exbuf);
  edge_msg<<<E_TOT, 256, 0, stream>>>(ei, nsum, exbuf, h_feat, acc);
  finalize_layer<<<fin_blocks, TPB, 0, stream>>>(acc, b1, h_out, ah);

  // ---------- layer 2 ----------
  transpose_pad_f16<<<(HFDIM * 64 + TPB - 1) / TPB, TPB, 0, stream>>>(W2, wt, 64, 64, HFDIM);
  wmma_gemm<64><<<gemm_blocks, 128, 0, stream>>>(ah, wt, h_feat, N_NODES);
  att_coef<<<att_blocks, TPB, 0, stream>>>(h_feat, as2, ad2, a_src, a_dst);
  hipMemsetAsync(nmax, 0, (size_t)N_NODES * HEADS * sizeof(unsigned), stream);
  hipMemsetAsync(nsum, 0, (size_t)N_NODES * HEADS * sizeof(float), stream);
  hipMemsetAsync(acc,  0, (size_t)N_NODES * HFDIM * sizeof(float), stream);
  edge_max<<<eh_blocks, TPB, 0, stream>>>(ei, a_src, a_dst, nmax);
  edge_sum<<<eh_blocks, TPB, 0, stream>>>(ei, a_src, a_dst, nmax, nsum, exbuf);
  edge_msg<<<E_TOT, 256, 0, stream>>>(ei, nsum, exbuf, h_feat, acc);
  finalize_layer<<<fin_blocks, TPB, 0, stream>>>(acc, b2, h_out, ah);

  // ---------- heads ----------
  start_mlp<<<NCAND, 64, 0, stream>>>(h_out, donor, cand, sW1, sb1, sW2, sb2, out);
  length_mlp<<<1, 128, 0, stream>>>(h_out, donor, cand, lW1, lb1, lW2, lb2, out);
}
